// RGINQM9_91182155694563
// MI455X (gfx1250) — matrix operand order repaired and checked by reading, not compiled, not run on hardware
//
#include <hip/hip_runtime.h>
#include <hip/hip_bf16.h>

// ---------------------------------------------------------------------------
// RGIN-QM9 for gfx1250 (MI455X).
// Heavy GEMMs: bf16 activations/weights, v_wmma_f32_16x16x32_bf16,
// double-buffered LDS tiles staged via GLOBAL_LOAD_ASYNC_TO_LDS_B128 with
// partial s_wait_asynccnt waits for load/compute overlap.
// ---------------------------------------------------------------------------

#define NN   50000
#define EE   300000
#define RR   4
#define FIN  11
#define HH   256
#define GG   2048
#define TT   19

typedef __attribute__((ext_vector_type(16))) __bf16 v16bf;
typedef __attribute__((ext_vector_type(8)))  float  v8f;

#if defined(__HIP_DEVICE_COMPILE__) && __has_builtin(__builtin_amdgcn_global_load_async_to_lds_b128)
#define USE_ASYNC_LDS 1
typedef int v4i __attribute__((vector_size(16)));      // GCC-vector int4 (matches builtin sig)
typedef __attribute__((address_space(1))) void as1_void;
typedef __attribute__((address_space(3))) void as3_void;
typedef __attribute__((address_space(1))) v4i  as1_v4i;
typedef __attribute__((address_space(3))) v4i  as3_v4i;
#define GPTR(p) ((as1_v4i*)(as1_void*)(void*)(p))
#define LPTR(p) ((as3_v4i*)(as3_void*)(void*)(p))
#endif

template <int Nw>
__device__ __forceinline__ void wait_async_lds() {
#if defined(USE_ASYNC_LDS)
#if __has_builtin(__builtin_amdgcn_s_wait_asynccnt)
    __builtin_amdgcn_s_wait_asynccnt(Nw);
#else
    if (Nw == 0) asm volatile("s_wait_asynccnt 0x0" ::: "memory");
    else         asm volatile("s_wait_asynccnt 0x3" ::: "memory");
#endif
#endif
}

// ======================= fast path: K = Ncols = 256, bf16 ===================
// C[M x 256] = A_bf16[M x 256] * BT_bf16[256 x 256]^T + bias (+C if acc)(relu)
// BT is pre-transposed: BT[n*256 + k] == B[k][n].
#define TS_M 128
#define TS_N 64
#define KSTEP 32
#define SA_STRIDE 40   // bf16 elems; 80 B rows -> every 16 B chunk aligned
#define SB_STRIDE 40
#define SA_ELEMS (TS_M * SA_STRIDE)
#define SB_ELEMS (TS_N * SB_STRIDE)

__global__ __launch_bounds__(256) void wmma_gemm_bf16(
    const __bf16* __restrict__ A, const __bf16* __restrict__ BT,
    const float* __restrict__ bias, float* __restrict__ C,
    int M, int acc, int relu)
{
    __shared__ __bf16 sA[2 * SA_ELEMS];
    __shared__ __bf16 sB[2 * SB_ELEMS];

    const int tid  = threadIdx.x;
    const int lane = tid & 31;
    const int wave = tid >> 5;
    const int half = (lane >> 4) & 1;
    const int l16  = lane & 15;

    const int row0 = blockIdx.y * TS_M;
    const int col0 = blockIdx.x * TS_N;
    const int wr = (wave >> 1) * 32;
    const int wc = (wave & 1)  * 32;

    // chunk coordinates (16 B = 8 bf16 per chunk, 4 chunks per 32-elem row)
    const int ar0 = (0 * 256 + tid) >> 2, ac0 = ((0 * 256 + tid) & 3) * 8;
    const int ar1 = (1 * 256 + tid) >> 2, ac1 = ((1 * 256 + tid) & 3) * 8;
    const int bn  = tid >> 2,             bc  = (tid & 3) * 8;

    // clamp rows instead of predicating: every thread issues exactly 3 async
    // loads per tile, so per-wave ASYNCcnt is deterministic for partial waits.
    const int gr0 = (row0 + ar0 < M) ? (row0 + ar0) : (M - 1);
    const int gr1 = (row0 + ar1 < M) ? (row0 + ar1) : (M - 1);

    auto stage_tile = [&](int k0, int buf) {
        __bf16* dA = sA + buf * SA_ELEMS;
        __bf16* dB = sB + buf * SB_ELEMS;
#if defined(USE_ASYNC_LDS)
        __builtin_amdgcn_global_load_async_to_lds_b128(
            GPTR(A + (long)gr0 * 256 + k0 + ac0), LPTR(&dA[ar0 * SA_STRIDE + ac0]), 0, 0);
        __builtin_amdgcn_global_load_async_to_lds_b128(
            GPTR(A + (long)gr1 * 256 + k0 + ac1), LPTR(&dA[ar1 * SA_STRIDE + ac1]), 0, 0);
        __builtin_amdgcn_global_load_async_to_lds_b128(
            GPTR(BT + (long)(col0 + bn) * 256 + k0 + bc), LPTR(&dB[bn * SB_STRIDE + bc]), 0, 0);
#else
        int4 va0 = *(const int4*)(A + (long)gr0 * 256 + k0 + ac0);
        int4 va1 = *(const int4*)(A + (long)gr1 * 256 + k0 + ac1);
        int4 vb  = *(const int4*)(BT + (long)(col0 + bn) * 256 + k0 + bc);
        *(int4*)(&dA[ar0 * SA_STRIDE + ac0]) = va0;
        *(int4*)(&dA[ar1 * SA_STRIDE + ac1]) = va1;
        *(int4*)(&dB[bn * SB_STRIDE + bc])   = vb;
#endif
    };

    v8f c00 = {}, c01 = {}, c10 = {}, c11 = {};

    stage_tile(0, 0);                              // prologue: tile 0 -> buf 0

    #pragma unroll
    for (int kt = 0; kt < 8; ++kt) {               // K = 256, step 32
        const int cur = kt & 1;
        if (kt < 7) {
            stage_tile((kt + 1) * KSTEP, cur ^ 1); // prefetch next tile
            wait_async_lds<3>();                   // only tile kt must be done
        } else {
            wait_async_lds<0>();
        }
        __syncthreads();                           // tile kt visible block-wide

        const __bf16* pA = sA + cur * SA_ELEMS;
        const __bf16* pB = sB + cur * SB_ELEMS;
        v16bf a0, a1, b0, b1;
        #pragma unroll
        for (int i = 0; i < 16; ++i) {
            int ka = i + (half ? 16 : 0);                      // A 16x32 layout
            a0[i] = pA[(wr +      l16) * SA_STRIDE + ka];
            a1[i] = pA[(wr + 16 + l16) * SA_STRIDE + ka];
            int kb = i + (half ? 16 : 0);                      // B 32x16 layout
            b0[i] = pB[(wc +      l16) * SB_STRIDE + kb];
            b1[i] = pB[(wc + 16 + l16) * SB_STRIDE + kb];
        }
        c00 = __builtin_amdgcn_wmma_f32_16x16x32_bf16(false, a0, false, b0, (short)0, c00, false, false);
        c01 = __builtin_amdgcn_wmma_f32_16x16x32_bf16(false, a0, false, b1, (short)0, c01, false, false);
        c10 = __builtin_amdgcn_wmma_f32_16x16x32_bf16(false, a1, false, b0, (short)0, c10, false, false);
        c11 = __builtin_amdgcn_wmma_f32_16x16x32_bf16(false, a1, false, b1, (short)0, c11, false, false);
        __syncthreads();                           // all reads of buf cur done
    }

    #pragma unroll
    for (int q = 0; q < 4; ++q) {
        v8f cc = (q == 0) ? c00 : (q == 1) ? c01 : (q == 2) ? c10 : c11;
        int rb = wr + ((q >= 2) ? 16 : 0);
        int cb = wc + ((q & 1) ? 16 : 0);
        #pragma unroll
        for (int j = 0; j < 8; ++j) {
            int r = row0 + rb + j + (half ? 8 : 0);
            int n = col0 + cb + l16;
            if (r < M) {
                float v = cc[j] + bias[n];
                if (acc)  v += C[(long)r * HH + n];
                if (relu) v = fmaxf(v, 0.0f);
                C[(long)r * HH + n] = v;
            }
        }
    }
}

// ================== small-K path (layer 1, K = 11, fp32 in) =================
#define SLDA 36
#define SLDB 36
__global__ __launch_bounds__(256) void wmma_gemm_smallk(
    const float* __restrict__ A, const float* __restrict__ A2,
    const float* __restrict__ B, const float* __restrict__ bias,
    float* __restrict__ C, int M, int K)
{
    __shared__ __bf16 sA[TS_M * SLDA];
    __shared__ __bf16 sB[TS_N * SLDB];

    const int tid  = threadIdx.x;
    const int lane = tid & 31;
    const int wave = tid >> 5;
    const int half = (lane >> 4) & 1;
    const int l16  = lane & 15;
    const int row0 = blockIdx.y * TS_M;
    const int col0 = blockIdx.x * TS_N;
    const int wr = (wave >> 1) * 32;
    const int wc = (wave & 1)  * 32;

    v8f c00 = {}, c01 = {}, c10 = {}, c11 = {};

    // single K tile (K <= 32), zero-padded
    #pragma unroll
    for (int i = 0; i < 16; ++i) {
        int idx = i * 256 + tid;
        int r = idx >> 5, k = idx & 31;
        int gr = row0 + r;
        float va = 0.0f;
        if (gr < M && k < K) {
            va = A[(long)gr * K + k];
            if (A2) va += A2[(long)gr * K + k];
        }
        sA[r * SLDA + k] = (__bf16)va;
    }
    #pragma unroll
    for (int i = 0; i < 8; ++i) {
        int idx = i * 256 + tid;
        int n = idx & 63, k = idx >> 6;
        float vb = 0.0f;
        if (k < K) vb = B[(long)k * HH + col0 + n];
        sB[n * SLDB + k] = (__bf16)vb;
    }
    __syncthreads();

    v16bf a0, a1, b0, b1;
    #pragma unroll
    for (int i = 0; i < 16; ++i) {
        int ka = i + (half ? 16 : 0);
        a0[i] = sA[(wr +      l16) * SLDA + ka];
        a1[i] = sA[(wr + 16 + l16) * SLDA + ka];
        int kb = i + (half ? 16 : 0);
        b0[i] = sB[(wc +      l16) * SLDB + kb];
        b1[i] = sB[(wc + 16 + l16) * SLDB + kb];
    }
    c00 = __builtin_amdgcn_wmma_f32_16x16x32_bf16(false, a0, false, b0, (short)0, c00, false, false);
    c01 = __builtin_amdgcn_wmma_f32_16x16x32_bf16(false, a0, false, b1, (short)0, c01, false, false);
    c10 = __builtin_amdgcn_wmma_f32_16x16x32_bf16(false, a1, false, b0, (short)0, c10, false, false);
    c11 = __builtin_amdgcn_wmma_f32_16x16x32_bf16(false, a1, false, b1, (short)0, c11, false, false);

    #pragma unroll
    for (int q = 0; q < 4; ++q) {
        v8f cc = (q == 0) ? c00 : (q == 1) ? c01 : (q == 2) ? c10 : c11;
        int rb = wr + ((q >= 2) ? 16 : 0);
        int cb = wc + ((q & 1) ? 16 : 0);
        #pragma unroll
        for (int j = 0; j < 8; ++j) {
            int r = row0 + rb + j + (half ? 8 : 0);
            int n = col0 + cb + l16;
            if (r < M)
                C[(long)r * HH + n] = cc[j] + bias[n];
        }
    }
}

// =========================== elementwise helpers ============================
__global__ __launch_bounds__(256) void convert_bf16(
    __bf16* __restrict__ y, const float* __restrict__ x, long n)
{
    long i = ((long)blockIdx.x * blockDim.x + threadIdx.x) * 4;
    if (i >= n) return;
    float4 v = *(const float4*)(x + i);
    __bf16* o = y + i;
    o[0] = (__bf16)v.x; o[1] = (__bf16)v.y; o[2] = (__bf16)v.z; o[3] = (__bf16)v.w;
}

__global__ __launch_bounds__(256) void convert_add_bf16(
    __bf16* __restrict__ y, const float* __restrict__ x,
    const float* __restrict__ a, long n)
{
    long i = ((long)blockIdx.x * blockDim.x + threadIdx.x) * 4;
    if (i >= n) return;
    float4 v = *(const float4*)(x + i);
    float4 w = *(const float4*)(a + i);
    __bf16* o = y + i;
    o[0] = (__bf16)(v.x + w.x); o[1] = (__bf16)(v.y + w.y);
    o[2] = (__bf16)(v.z + w.z); o[3] = (__bf16)(v.w + w.w);
}

// wT_bf16[n*256 + k] = w[k*256 + n]
__global__ __launch_bounds__(256) void convert_transpose_w(
    __bf16* __restrict__ wT, const float* __restrict__ w)
{
    int i = blockIdx.x * blockDim.x + threadIdx.x;   // 65536 threads
    int n = i & 255, k = i >> 8;
    wT[n * 256 + k] = (__bf16)w[k * 256 + n];
}

// ================================ graph ops =================================
__global__ __launch_bounds__(256) void scatter_edges(
    const float* __restrict__ x, const int* __restrict__ ei,
    const int* __restrict__ et, float* __restrict__ agg, int F, int rel)
{
    int e = blockIdx.x * 8 + (threadIdx.x >> 5);
    if (e >= EE) return;
    if (et[e] != rel) return;
    int src = ei[e];
    int dst = ei[EE + e];
    int lane = threadIdx.x & 31;
    for (int f = lane; f < F; f += 32)
        atomicAdd(&agg[(long)dst * F + f], x[(long)src * F + f]);
}

__global__ __launch_bounds__(256) void bn_partial(
    const float* __restrict__ h, float* __restrict__ sums,
    float* __restrict__ sumsq, int M)
{
    int c = threadIdx.x;
    float s = 0.0f, s2 = 0.0f;
    for (int r = blockIdx.x; r < M; r += gridDim.x) {
        float v = h[(long)r * HH + c];
        s += v; s2 += v * v;
    }
    atomicAdd(&sums[c], s);
    atomicAdd(&sumsq[c], s2);
}

__global__ void bn_finalize(const float* __restrict__ sums, const float* __restrict__ sumsq,
                            const float* __restrict__ g, const float* __restrict__ bt,
                            float* __restrict__ scale, float* __restrict__ shift, int M)
{
    int c = threadIdx.x;
    float mu  = sums[c] / (float)M;
    float var = sumsq[c] / (float)M - mu * mu;
    float rs  = rsqrtf(var + 1e-5f);
    float sc  = g[c] * rs;
    scale[c] = sc;
    shift[c] = bt[c] - mu * sc;
}

// h_bf16 = relu(h*scale + shift), bf16 out for the w2 GEMM
__global__ __launch_bounds__(256) void bn_apply_relu_bf16(
    const float* __restrict__ h, const float* __restrict__ scale,
    const float* __restrict__ shift, __bf16* __restrict__ out)
{
    long i = (long)blockIdx.x * blockDim.x + threadIdx.x;
    if (i >= (long)NN * HH) return;
    int c = (int)(i & (HH - 1));
    out[i] = (__bf16)fmaxf(h[i] * scale[c] + shift[c], 0.0f);
}

__global__ __launch_bounds__(256) void pool_kernel(
    const float* __restrict__ h, const int* __restrict__ batch,
    float* __restrict__ pooled, float* __restrict__ counts)
{
    long i = (long)blockIdx.x * blockDim.x + threadIdx.x;
    if (i >= (long)NN * HH) return;
    int n = (int)(i >> 8);
    int f = (int)(i & (HH - 1));
    int g = batch[n];
    atomicAdd(&pooled[(long)g * HH + f], h[i]);
    if (f == 0) atomicAdd(&counts[g], 1.0f);
}

__global__ __launch_bounds__(256) void final_linear(
    const float* __restrict__ pooled, const float* __restrict__ counts,
    const float* __restrict__ W, const float* __restrict__ b,
    float* __restrict__ out)
{
    int i = blockIdx.x * blockDim.x + threadIdx.x;
    if (i >= GG * TT) return;
    int g = i / TT, t = i % TT;
    float inv = 1.0f / fmaxf(counts[g], 1.0f);
    float s = 0.0f;
    for (int k = 0; k < HH; ++k)
        s += pooled[(long)g * HH + k] * inv * W[k * TT + t];
    out[i] = s + b[t];
}

// ---------------------------------------------------------------------------
struct Scratch {
    float *bufA, *bufB, *agg, *hbuf;
    __bf16 *xb, *tmpb, *hb;
    float *sums, *sumsq, *scale, *shift, *pooled, *counts;
};

static void bn_block(const float* hbuf, const float* g, const float* bt, int r,
                     Scratch& S, hipStream_t stream)
{
    (void)hipMemsetAsync(S.sums,  0, HH * sizeof(float), stream);
    (void)hipMemsetAsync(S.sumsq, 0, HH * sizeof(float), stream);
    bn_partial<<<400, 256, 0, stream>>>(hbuf, S.sums, S.sumsq, NN);
    bn_finalize<<<1, 256, 0, stream>>>(S.sums, S.sumsq, g + (size_t)r * HH,
                                       bt + (size_t)r * HH, S.scale, S.shift, NN);
    bn_apply_relu_bf16<<<((long)NN * HH + 255) / 256, 256, 0, stream>>>(
        hbuf, S.scale, S.shift, S.hb);
}

extern "C" void kernel_launch(void* const* d_in, const int* in_sizes, int n_in,
                              void* d_out, int out_size, void* d_ws, size_t ws_size,
                              hipStream_t stream)
{
    const float* x     = (const float*)d_in[0];
    const int*   ei    = (const int*)d_in[1];
    const int*   et    = (const int*)d_in[2];
    const int*   batch = (const int*)d_in[3];
    const float* L[3][8];   // sw, sb, w1, b1, g, bt, w2, b2
    for (int li = 0; li < 3; ++li)
        for (int w = 0; w < 8; ++w)
            L[li][w] = (const float*)d_in[4 + li * 8 + w];
    const float* lin_w = (const float*)d_in[28];
    const float* lin_b = (const float*)d_in[29];

    char* ws = (char*)d_ws;
    auto carve = [&](size_t bytes) {
        void* p = (void*)ws;
        ws += (bytes + 255) & ~(size_t)255;
        return p;
    };
    Scratch S;
    S.bufA   = (float*)carve((size_t)NN * HH * sizeof(float));
    S.bufB   = (float*)carve((size_t)NN * HH * sizeof(float));
    S.agg    = (float*)carve((size_t)NN * HH * sizeof(float));
    S.hbuf   = (float*)carve((size_t)NN * HH * sizeof(float));
    S.xb     = (__bf16*)carve((size_t)NN * HH * sizeof(__bf16));
    S.tmpb   = (__bf16*)carve((size_t)NN * HH * sizeof(__bf16));
    S.hb     = (__bf16*)carve((size_t)NN * HH * sizeof(__bf16));
    S.sums   = (float*)carve(HH * sizeof(float));
    S.sumsq  = (float*)carve(HH * sizeof(float));
    S.scale  = (float*)carve(HH * sizeof(float));
    S.shift  = (float*)carve(HH * sizeof(float));
    S.pooled = (float*)carve((size_t)GG * HH * sizeof(float));
    S.counts = (float*)carve(GG * sizeof(float));
    // bf16 transposed weights: l1_w2[4] | l2: sw, w1[4], w2[4] | l3: sw, w1[4], w2[4]
    __bf16* wT = (__bf16*)carve((size_t)22 * 65536 * sizeof(__bf16));
    __bf16* l1w2T = wT;
    __bf16* swT[3]  = {nullptr, wT + 4 * 65536,  wT + 13 * 65536};
    __bf16* w1T[3]  = {nullptr, wT + 5 * 65536,  wT + 14 * 65536};
    __bf16* w2T[3]  = {l1w2T,   wT + 9 * 65536,  wT + 18 * 65536};

    // ---- weight conversion (+transpose) ----
    for (int r = 0; r < RR; ++r)
        convert_transpose_w<<<256, 256, 0, stream>>>(l1w2T + (size_t)r * 65536,
                                                     L[0][6] + (size_t)r * 65536);
    for (int li = 1; li < 3; ++li) {
        convert_transpose_w<<<256, 256, 0, stream>>>(swT[li], L[li][0]);
        for (int r = 0; r < RR; ++r) {
            convert_transpose_w<<<256, 256, 0, stream>>>(w1T[li] + (size_t)r * 65536,
                                                         L[li][2] + (size_t)r * 65536);
            convert_transpose_w<<<256, 256, 0, stream>>>(w2T[li] + (size_t)r * 65536,
                                                         L[li][6] + (size_t)r * 65536);
        }
    }

    dim3 gblk(HH / TS_N, (NN + TS_M - 1) / TS_M);
    long cvt_blocks = ((long)NN * HH / 4 + 255) / 256;

    // =============================== layer 1 ================================
    wmma_gemm_smallk<<<gblk, 256, 0, stream>>>(x, nullptr, L[0][0], L[0][1],
                                               S.bufA, NN, FIN);
    for (int r = 0; r < RR; ++r) {
        (void)hipMemsetAsync(S.agg, 0, (size_t)NN * FIN * sizeof(float), stream);
        scatter_edges<<<(EE + 7) / 8, 256, 0, stream>>>(x, ei, et, S.agg, FIN, r);
        wmma_gemm_smallk<<<gblk, 256, 0, stream>>>(x, S.agg,
                                                   L[0][2] + (size_t)r * FIN * HH,
                                                   L[0][3] + (size_t)r * HH,
                                                   S.hbuf, NN, FIN);
        bn_block(S.hbuf, L[0][4], L[0][5], r, S, stream);
        wmma_gemm_bf16<<<gblk, 256, 0, stream>>>(S.hb, l1w2T + (size_t)r * 65536,
                                                 L[0][7] + (size_t)r * HH,
                                                 S.bufA, NN, 1, (r == RR - 1) ? 1 : 0);
    }

    // ============================ layers 2 & 3 ==============================
    float* lay_in  = S.bufA;
    float* lay_out = S.bufB;
    for (int li = 1; li < 3; ++li) {
        convert_bf16<<<cvt_blocks, 256, 0, stream>>>(S.xb, lay_in, (long)NN * HH);
        wmma_gemm_bf16<<<gblk, 256, 0, stream>>>(S.xb, swT[li], L[li][1],
                                                 lay_out, NN, 0, 0);
        for (int r = 0; r < RR; ++r) {
            (void)hipMemsetAsync(S.agg, 0, (size_t)NN * HH * sizeof(float), stream);
            scatter_edges<<<(EE + 7) / 8, 256, 0, stream>>>(lay_in, ei, et, S.agg, HH, r);
            convert_add_bf16<<<cvt_blocks, 256, 0, stream>>>(S.tmpb, lay_in, S.agg,
                                                             (long)NN * HH);
            wmma_gemm_bf16<<<gblk, 256, 0, stream>>>(S.tmpb, w1T[li] + (size_t)r * 65536,
                                                     L[li][3] + (size_t)r * HH,
                                                     S.hbuf, NN, 0, 0);
            bn_block(S.hbuf, L[li][4], L[li][5], r, S, stream);
            wmma_gemm_bf16<<<gblk, 256, 0, stream>>>(S.hb, w2T[li] + (size_t)r * 65536,
                                                     L[li][7] + (size_t)r * HH,
                                                     lay_out, NN, 1, (r == RR - 1) ? 1 : 0);
        }
        float* t = lay_in; lay_in = lay_out; lay_out = t;
    }
    // after loop: final layer output is in lay_in

    (void)hipMemsetAsync(S.pooled, 0, (size_t)GG * HH * sizeof(float), stream);
    (void)hipMemsetAsync(S.counts, 0, GG * sizeof(float), stream);
    pool_kernel<<<((long)NN * HH + 255) / 256, 256, 0, stream>>>(lay_in, batch,
                                                                 S.pooled, S.counts);
    final_linear<<<(GG * TT + 255) / 256, 256, 0, stream>>>(S.pooled, S.counts,
                                                            lin_w, lin_b, (float*)d_out);
}
